// ResonanceGatedAttention_67353677136474
// MI455X (gfx1250) — matrix operand order repaired and checked by reading, not compiled, not run on hardware
//
#include <hip/hip_runtime.h>
#include <hip/hip_bf16.h>

// ---------------------------------------------------------------------------
// ResonanceGatedAttention for MI455X (gfx1250, wave32, WMMA f32<=f16xf16)
// ---------------------------------------------------------------------------
typedef __attribute__((ext_vector_type(16))) _Float16 v16h;
typedef __attribute__((ext_vector_type(8)))  _Float16 v8h;
typedef __attribute__((ext_vector_type(8)))  float    v8f;
typedef __attribute__((ext_vector_type(4)))  unsigned int u32x4;
typedef __attribute__((ext_vector_type(8)))  int          i32x8;
typedef __attribute__((ext_vector_type(4)))  int          i32x4;

#define D_MODEL 1024
#define N_HEADS 16
#define D_HEAD  64
#define SEQ     2048
#define BATCH   2
#define NTOK    (BATCH * SEQ)

#if __has_builtin(__builtin_amdgcn_tensor_load_to_lds)
#define HAVE_TDM 1
#else
#define HAVE_TDM 0
#endif

__device__ __forceinline__ v8f wmma_f16(v16h a, v16h b, v8f c) {
  // D = A(16x32 f16) * B(32x16 f16) + C(16x16 f32)
  return __builtin_amdgcn_wmma_f32_16x16x32_f16(false, a, false, b, (short)0, c,
                                                false, false);
}

__device__ __forceinline__ v8f vzero8() {
  v8f v;
#pragma unroll
  for (int i = 0; i < 8; ++i) v[i] = 0.f;
  return v;
}

// A operand (16x32, 16-bit, row-major with leading dim ld):
// lane L: M=L&15, K={koff..koff+7, koff+16..koff+23}, koff = 8*(L>=16).
__device__ __forceinline__ v16h load_a_16x32(const _Float16* base, int ld, int lane) {
  const int m    = lane & 15;
  const int koff = (lane >> 4) << 3;
  const _Float16* p = base + (size_t)m * ld + koff;
  v8h lo = *(const v8h*)p;         // K = koff .. koff+7
  v8h hi = *(const v8h*)(p + 16);  // K = koff+16 .. koff+23
  v16h a;
#pragma unroll
  for (int i = 0; i < 8; ++i) { a[i] = lo[i]; a[i + 8] = hi[i]; }
  return a;
}

// B operand (32x16, 16-bit) from memory stored "transposed-contiguous":
// element (k, n) = bt[n*ld + k]. Lane L: col L&15, K={koff..koff+7, koff+16..koff+23}, koff = 8*(L>=16).
__device__ __forceinline__ v16h load_bT_32x16(const _Float16* bt, int ld, int lane) {
  const int n    = lane & 15;
  const int koff = (lane >> 4) << 3;
  const _Float16* p = bt + (size_t)n * ld + koff;
  v8h lo = *(const v8h*)p;        // k = koff .. koff+7
  v8h hi = *(const v8h*)(p + 16); // k = koff+16 .. koff+23
  v16h b;
#pragma unroll
  for (int i = 0; i < 8; ++i) { b[i] = lo[i]; b[i + 8] = hi[i]; }
  return b;
}

#if HAVE_TDM
// Tensor Data Mover: DMA a 2D tile (rows x rowlen f16 elements, row stride in
// elements) from global memory into LDS at byte offset lds_off.
// Descriptor packing per CDNA5 ISA.
// This toolchain's builtin is the 6-arg form:
//   (uint32x4 g0, int32x8 g1, int32x4 g2, int32x4 g3, int32x8, int cpol)
__device__ __forceinline__ void tdm_load_2d_f16(uint32_t lds_off, const void* gptr,
                                                int tile_w, int tile_h,
                                                int tensor_w, int tensor_h,
                                                int row_stride) {
  const unsigned long long ga = (unsigned long long)(size_t)gptr;
  u32x4 g0;
  g0[0] = 1u;                                                   // count=1 (valid user D#)
  g0[1] = lds_off;                                              // lds_addr (bytes)
  g0[2] = (unsigned int)ga;                                     // global_addr[31:0]
  g0[3] = ((unsigned int)(ga >> 32) & 0x01FFFFFFu) | 0x80000000u; // addr[56:32] | type=2
  i32x8 g1;
  g1[0] = 1 << 16;                                              // data_size=1 -> 2 bytes
  g1[1] = (tensor_w & 0xFFFF) << 16;                            // tensor_dim0[15:0]
  g1[2] = ((tensor_w >> 16) & 0xFFFF) | ((tensor_h & 0xFFFF) << 16); // dim0 hi | dim1 lo
  g1[3] = ((tensor_h >> 16) & 0xFFFF) | ((tile_w & 0xFFFF) << 16);   // dim1 hi | tile_dim0
  g1[4] = (tile_h & 0xFFFF);                                    // tile_dim1 (tile_dim2=0)
  g1[5] = row_stride;                                           // tensor_dim0_stride lo32
  g1[6] = 0;                                                    // stride0 hi | stride1 lo
  g1[7] = 0;
  i32x4 gz4;
  gz4[0] = 0; gz4[1] = 0; gz4[2] = 0; gz4[3] = 0;
  i32x8 gz8;
#pragma unroll
  for (int i = 0; i < 8; ++i) gz8[i] = 0;
  __builtin_amdgcn_tensor_load_to_lds(g0, g1, gz4, gz4, gz8, 0);
}
#endif

// ---------------------------------------------------------------------------
// K0: f32 -> f16 weight conversion
// ---------------------------------------------------------------------------
__global__ void cvt_f32_f16(const float* __restrict__ s, _Float16* __restrict__ d, int n) {
  int i = (blockIdx.x * blockDim.x + threadIdx.x) * 4;
  if (i + 3 < n) {
#pragma unroll
    for (int j = 0; j < 4; ++j) d[i + j] = (_Float16)s[i + j];
  }
}

// ---------------------------------------------------------------------------
// K1: fused LayerNorm + phase = xn . Wp ; writes xn as f16. One block / token.
// ---------------------------------------------------------------------------
__global__ __launch_bounds__(256) void ln_phase_kernel(
    const float* __restrict__ x, const float* __restrict__ gamma,
    const float* __restrict__ beta, const float* __restrict__ Wp,
    _Float16* __restrict__ xnf, float* __restrict__ phase,
    float* __restrict__ energy) {
  __shared__ float red[256];
  const int tok = blockIdx.x;
  const int tid = threadIdx.x;
  const float* xr = x + (size_t)tok * D_MODEL;

  float lx[4];
  float s = 0.f;
#pragma unroll
  for (int i = 0; i < 4; ++i) { lx[i] = xr[tid + i * 256]; s += lx[i]; }
  red[tid] = s; __syncthreads();
  for (int st = 128; st > 0; st >>= 1) { if (tid < st) red[tid] += red[tid + st]; __syncthreads(); }
  const float mu = red[0] * (1.f / D_MODEL);
  __syncthreads();

  s = 0.f;
#pragma unroll
  for (int i = 0; i < 4; ++i) { float d0 = lx[i] - mu; s += d0 * d0; }
  red[tid] = s; __syncthreads();
  for (int st = 128; st > 0; st >>= 1) { if (tid < st) red[tid] += red[tid + st]; __syncthreads(); }
  const float rstd = rsqrtf(red[0] * (1.f / D_MODEL) + 1e-5f);
  __syncthreads();

  float ps = 0.f;
#pragma unroll
  for (int i = 0; i < 4; ++i) {
    const int d = tid + i * 256;
    const float xn = (lx[i] - mu) * rstd * gamma[d] + beta[d];
    xnf[(size_t)tok * D_MODEL + d] = (_Float16)xn;
    ps += xn * Wp[d];
  }
  red[tid] = ps; __syncthreads();
  for (int st = 128; st > 0; st >>= 1) { if (tid < st) red[tid] += red[tid + st]; __syncthreads(); }
  if (tid == 0) phase[tok] = red[0];
  if (tok == 0 && tid == 0) *energy = 0.f;   // zeroed every launch, before attn atomics
}

// ---------------------------------------------------------------------------
// K2: out[m,n] = scale * sum_k A[m,k] * W[n,k]   (torch Linear, no bias)
// Register-blocked: one wave -> 32x64 tile = 2 A ops x 4 B ops -> 8 WMMA/K-step.
// ---------------------------------------------------------------------------
__global__ __launch_bounds__(128) void proj_f16(
    const _Float16* __restrict__ A, const _Float16* __restrict__ W,
    _Float16* __restrict__ out, float scale) {
  const int lane = threadIdx.x & 31;
  const int wv   = threadIdx.x >> 5;
  const int n0   = (blockIdx.x * 4 + wv) * 64;   // 64 output cols per wave
  const int m0   = blockIdx.y * 32;              // 32 output rows per wave

  v8f c[2][4];
#pragma unroll
  for (int i2 = 0; i2 < 2; ++i2)
#pragma unroll
    for (int j = 0; j < 4; ++j) c[i2][j] = vzero8();

  for (int k0 = 0; k0 < D_MODEL; k0 += 32) {
    v16h a0 = load_a_16x32(A + (size_t)m0 * D_MODEL + k0, D_MODEL, lane);
    v16h a1 = load_a_16x32(A + (size_t)(m0 + 16) * D_MODEL + k0, D_MODEL, lane);
#pragma unroll
    for (int j = 0; j < 4; ++j) {
      v16h b = load_bT_32x16(W + (size_t)(n0 + 16 * j) * D_MODEL + k0, D_MODEL, lane);
      c[0][j] = wmma_f16(a0, b, c[0][j]);
      c[1][j] = wmma_f16(a1, b, c[1][j]);
    }
  }
  const int hf = lane >> 4, cn = lane & 15;
#pragma unroll
  for (int i2 = 0; i2 < 2; ++i2)
#pragma unroll
    for (int j = 0; j < 4; ++j)
#pragma unroll
      for (int r = 0; r < 8; ++r)
        out[(size_t)(m0 + 16 * i2 + r + 8 * hf) * D_MODEL + n0 + 16 * j + cn] =
            (_Float16)(c[i2][j][r] * scale);
}

// ---------------------------------------------------------------------------
// K3: flash attention + resonance bias + carrier gating + |out| energy.
// Block = (64 query rows, one (b,h)); 4 waves, each owns a 16-row tile.
// K tile is DMA'd once per block via the Tensor Data Mover (TENSORcnt),
// removing the 4x redundant per-wave global fetch of the shared K tile.
// ---------------------------------------------------------------------------
__global__ __launch_bounds__(128) void attn_kernel(
    const _Float16* __restrict__ qf, const _Float16* __restrict__ kf,
    const _Float16* __restrict__ vf, const float* __restrict__ phase,
    const float* __restrict__ carrier, const float* __restrict__ lamp,
    _Float16* __restrict__ aof, float* __restrict__ energy) {
  __shared__ __align__(16) _Float16 kT[32 * D_HEAD];      // TDM-staged K tile [j][d]
  __shared__ __align__(16) _Float16 vT[D_HEAD * 32];      // staged V tile, [d][j]
  __shared__ __align__(16) _Float16 Pb[4][16 * 32];       // per-wave P bounce

  const int lane = threadIdx.x & 31;
  const int wv   = threadIdx.x >> 5;
  const int bb   = blockIdx.z;
  const int h    = blockIdx.y;
  const int i0   = blockIdx.x * 64 + wv * 16;
  const int hf   = lane >> 4, cn = lane & 15;

  const float lam = lamp[0];
  const size_t bD = (size_t)bb * SEQ * D_MODEL;
  const _Float16* qb = qf + bD + (size_t)i0 * D_MODEL + h * D_HEAD;
  const _Float16* kb = kf + bD + h * D_HEAD;
  const _Float16* vb = vf + bD + h * D_HEAD;
  const float*    ph = phase + (size_t)bb * SEQ;

  // Query tile held in registers for the whole row sweep (scale pre-folded).
  v16h aq0 = load_a_16x32(qb, D_MODEL, lane);
  v16h aq1 = load_a_16x32(qb + 32, D_MODEL, lane);

  float phi[8];
#pragma unroll
  for (int r = 0; r < 8; ++r) phi[r] = ph[i0 + r + 8 * hf];

  float mrow[8], lrow[8];
#pragma unroll
  for (int r = 0; r < 8; ++r) { mrow[r] = -3.0e38f; lrow[r] = 0.f; }
  v8f acc[4];
#pragma unroll
  for (int t = 0; t < 4; ++t) acc[t] = vzero8();

#if HAVE_TDM
  const uint32_t kT_off = (uint32_t)(size_t)&kT[0];  // LDS byte offset
#endif

  for (int jt = 0; jt < SEQ / 32; ++jt) {
    const int j0 = jt * 32;
    __syncthreads();  // previous iteration's kT/vT reads are done
#if HAVE_TDM
    if (wv == 0)      // one wave DMAs the shared 32x64 K tile into LDS
      tdm_load_2d_f16(kT_off, kb + (size_t)j0 * D_MODEL,
                      /*tile_w=*/D_HEAD, /*tile_h=*/32,
                      /*tensor_w=*/D_HEAD, /*tensor_h=*/SEQ,
                      /*row_stride=*/D_MODEL);
#endif
    {   // cooperative stage of V tile transposed: vT[d][jloc]
      const int jl = threadIdx.x >> 2;          // 0..31
      const int d0 = (threadIdx.x & 3) * 16;    // 0,16,32,48
      const _Float16* src = vb + (size_t)(j0 + jl) * D_MODEL + d0;
      v8h x0 = *(const v8h*)src;
      v8h x1 = *(const v8h*)(src + 8);
#pragma unroll
      for (int i = 0; i < 8; ++i) {
        vT[(d0 + i) * 32 + jl]     = x0[i];
        vT[(d0 + 8 + i) * 32 + jl] = x1[i];
      }
    }
    // one j-tile of lookahead into WGP$/L2 (global_prefetch_b8)
    if (jt + 1 < SEQ / 32) {
      __builtin_prefetch(kb + (size_t)(j0 + 32 + lane) * D_MODEL, 0, 1);
      __builtin_prefetch(vb + (size_t)(j0 + 32 + lane) * D_MODEL, 0, 1);
    }
#if HAVE_TDM
    if (wv == 0) __builtin_amdgcn_s_wait_tensorcnt(0);
    __syncthreads();  // kT (TDM) + vT staged and visible to all waves
#endif

    // ---- scores: two 16x16 C tiles (cols j0..j0+15, j0+16..j0+31) ----
    v8f s0 = vzero8(), s1 = vzero8();
#if HAVE_TDM
    s0 = wmma_f16(aq0, load_bT_32x16(kT,                 D_HEAD, lane), s0);
    s0 = wmma_f16(aq1, load_bT_32x16(kT + 32,            D_HEAD, lane), s0);
    s1 = wmma_f16(aq0, load_bT_32x16(kT + 16 * D_HEAD,      D_HEAD, lane), s1);
    s1 = wmma_f16(aq1, load_bT_32x16(kT + 16 * D_HEAD + 32, D_HEAD, lane), s1);
#else
    s0 = wmma_f16(aq0, load_bT_32x16(kb + (size_t)j0 * D_MODEL,        D_MODEL, lane), s0);
    s0 = wmma_f16(aq1, load_bT_32x16(kb + (size_t)j0 * D_MODEL + 32,   D_MODEL, lane), s0);
    s1 = wmma_f16(aq0, load_bT_32x16(kb + (size_t)(j0+16) * D_MODEL,      D_MODEL, lane), s1);
    s1 = wmma_f16(aq1, load_bT_32x16(kb + (size_t)(j0+16) * D_MODEL + 32, D_MODEL, lane), s1);
#endif

    // resonance bias: lam * cos^2((phi_i - phi_j)/2)
    const float pj0 = ph[j0 + cn], pj1 = ph[j0 + 16 + cn];
#pragma unroll
    for (int r = 0; r < 8; ++r) {
      float c0 = __cosf((phi[r] - pj0) * 0.5f); s0[r] += lam * c0 * c0;
      float c1 = __cosf((phi[r] - pj1) * 0.5f); s1[r] += lam * c1 * c1;
    }

    // ---- online softmax (rows live across 16-lane halves in one VGPR) ----
    float p0[8], p1[8];
#pragma unroll
    for (int r = 0; r < 8; ++r) {
      float t = fmaxf(s0[r], s1[r]);
      t = fmaxf(t, __shfl_xor(t, 1, 32));
      t = fmaxf(t, __shfl_xor(t, 2, 32));
      t = fmaxf(t, __shfl_xor(t, 4, 32));
      t = fmaxf(t, __shfl_xor(t, 8, 32));
      const float mn   = fmaxf(mrow[r], t);
      const float corr = __expf(mrow[r] - mn);
      mrow[r] = mn;
      p0[r] = __expf(s0[r] - mn);
      p1[r] = __expf(s1[r] - mn);
      float rs = p0[r] + p1[r];
      rs += __shfl_xor(rs, 1, 32);
      rs += __shfl_xor(rs, 2, 32);
      rs += __shfl_xor(rs, 4, 32);
      rs += __shfl_xor(rs, 8, 32);
      lrow[r] = lrow[r] * corr + rs;
#pragma unroll
      for (int t4 = 0; t4 < 4; ++t4) acc[t4][r] *= corr;
    }

    // ---- P: C layout -> A layout via per-wave LDS bounce ----
    _Float16* pbw = Pb[wv];
#pragma unroll
    for (int r = 0; r < 8; ++r) {
      pbw[(r + 8 * hf) * 32 + cn]      = (_Float16)p0[r];
      pbw[(r + 8 * hf) * 32 + cn + 16] = (_Float16)p1[r];
    }
    __syncthreads();  // Pb visible (and, in fallback path, vT staged)

    v16h ap;
    {
      const v8h* rowp = (const v8h*)(pbw + (size_t)cn * 32);
      v8h lo = rowp[hf];       // K = 8*hf .. 8*hf+7
      v8h hi = rowp[hf + 2];   // K = 8*hf+16 .. +23
#pragma unroll
      for (int i = 0; i < 8; ++i) { ap[i] = lo[i]; ap[i + 8] = hi[i]; }
    }
#pragma unroll
    for (int t4 = 0; t4 < 4; ++t4)
      acc[t4] = wmma_f16(ap, load_bT_32x16(vT + t4 * 16 * 32, 32, lane), acc[t4]);
  }

  // ---- epilogue: 1/l, carrier gate, energy, f16 store ----
  const float carr = carrier[h];
  float e = 0.f;
#pragma unroll
  for (int r = 0; r < 8; ++r) {
    const float inv = 1.f / lrow[r];
    float g = __cosf((phi[r] - carr) * 0.5f);
    const float sc = inv * g * g;
    const int row = i0 + r + 8 * hf;
#pragma unroll
    for (int t4 = 0; t4 < 4; ++t4) {
      const float o = acc[t4][r] * sc;
      e += fabsf(o);
      aof[bD + (size_t)row * D_MODEL + h * D_HEAD + t4 * 16 + cn] = (_Float16)o;
    }
  }
  e += __shfl_xor(e, 16, 32);
  e += __shfl_xor(e, 8, 32);
  e += __shfl_xor(e, 4, 32);
  e += __shfl_xor(e, 2, 32);
  e += __shfl_xor(e, 1, 32);
  if (lane == 0) atomicAdd(energy, e);
}

// ---------------------------------------------------------------------------
// K4: y[m,n] = resid[m,n] + sum_e A[m,e] * Wo[n,e]   (f32 out, same blocking)
// ---------------------------------------------------------------------------
__global__ __launch_bounds__(128) void out_proj_kernel(
    const _Float16* __restrict__ A, const _Float16* __restrict__ W,
    const float* __restrict__ resid, float* __restrict__ y) {
  const int lane = threadIdx.x & 31;
  const int wv   = threadIdx.x >> 5;
  const int n0   = (blockIdx.x * 4 + wv) * 64;
  const int m0   = blockIdx.y * 32;

  v8f c[2][4];
#pragma unroll
  for (int i2 = 0; i2 < 2; ++i2)
#pragma unroll
    for (int j = 0; j < 4; ++j) c[i2][j] = vzero8();

  for (int k0 = 0; k0 < D_MODEL; k0 += 32) {
    v16h a0 = load_a_16x32(A + (size_t)m0 * D_MODEL + k0, D_MODEL, lane);
    v16h a1 = load_a_16x32(A + (size_t)(m0 + 16) * D_MODEL + k0, D_MODEL, lane);
#pragma unroll
    for (int j = 0; j < 4; ++j) {
      v16h b = load_bT_32x16(W + (size_t)(n0 + 16 * j) * D_MODEL + k0, D_MODEL, lane);
      c[0][j] = wmma_f16(a0, b, c[0][j]);
      c[1][j] = wmma_f16(a1, b, c[1][j]);
    }
  }
  const int hf = lane >> 4, cn = lane & 15;
#pragma unroll
  for (int i2 = 0; i2 < 2; ++i2)
#pragma unroll
    for (int j = 0; j < 4; ++j)
#pragma unroll
      for (int r = 0; r < 8; ++r) {
        const size_t idx =
            (size_t)(m0 + 16 * i2 + r + 8 * hf) * D_MODEL + n0 + 16 * j + cn;
        y[idx] = resid[idx] + c[i2][j][r];
      }
}

// ---------------------------------------------------------------------------
extern "C" void kernel_launch(void* const* d_in, const int* in_sizes, int n_in,
                              void* d_out, int out_size, void* d_ws, size_t ws_size,
                              hipStream_t stream) {
  const float* x       = (const float*)d_in[0];
  const float* Wq      = (const float*)d_in[1];
  const float* Wk      = (const float*)d_in[2];
  const float* Wv      = (const float*)d_in[3];
  const float* Wo      = (const float*)d_in[4];
  const float* Wp      = (const float*)d_in[5];
  const float* gamma   = (const float*)d_in[6];
  const float* beta    = (const float*)d_in[7];
  const float* carrier = (const float*)d_in[8];
  const float* lam     = (const float*)d_in[9];

  float* y      = (float*)d_out;
  float* energy = y + (size_t)NTOK * D_MODEL;   // last element of d_out

  // workspace layout
  const size_t MB = 1024ull * 1024ull;
  char* ws = (char*)d_ws;
  _Float16* xnf  = (_Float16*)(ws + 0 * MB);
  _Float16* qf   = (_Float16*)(ws + 8 * MB);
  _Float16* kf   = (_Float16*)(ws + 16 * MB);
  _Float16* vf   = (_Float16*)(ws + 24 * MB);
  _Float16* aof  = (_Float16*)(ws + 32 * MB);
  _Float16* wq16 = (_Float16*)(ws + 40 * MB);
  _Float16* wk16 = (_Float16*)(ws + 42 * MB);
  _Float16* wv16 = (_Float16*)(ws + 44 * MB);
  _Float16* wo16 = (_Float16*)(ws + 46 * MB);
  float*    phase = (float*)(ws + 48 * MB);

  const int nW = D_MODEL * D_MODEL;
  cvt_f32_f16<<<nW / (256 * 4), 256, 0, stream>>>(Wq, wq16, nW);
  cvt_f32_f16<<<nW / (256 * 4), 256, 0, stream>>>(Wk, wk16, nW);
  cvt_f32_f16<<<nW / (256 * 4), 256, 0, stream>>>(Wv, wv16, nW);
  cvt_f32_f16<<<nW / (256 * 4), 256, 0, stream>>>(Wo, wo16, nW);

  ln_phase_kernel<<<NTOK, 256, 0, stream>>>(x, gamma, beta, Wp, xnf, phase, energy);

  // 32x64 per wave, 4 waves/block: grid = (1024/(4*64), 4096/32) = (4, 128)
  dim3 gProj(D_MODEL / 256, NTOK / 32);
  proj_f16<<<gProj, 128, 0, stream>>>(xnf, wq16, qf, 0.125f);  // 1/sqrt(64) folded
  proj_f16<<<gProj, 128, 0, stream>>>(xnf, wk16, kf, 1.0f);
  proj_f16<<<gProj, 128, 0, stream>>>(xnf, wv16, vf, 1.0f);

  attn_kernel<<<dim3(SEQ / 64, N_HEADS, BATCH), 128, 0, stream>>>(
      qf, kf, vf, phase, carrier, lam, aof, energy);

  out_proj_kernel<<<gProj, 128, 0, stream>>>(aof, wo16, x, y);
}
